// ConvLayer_12575664243444
// MI455X (gfx1250) — matrix operand order repaired and checked by reading, not compiled, not run on hardware
//
#include <hip/hip_runtime.h>

typedef _Float16 h16;
typedef __attribute__((ext_vector_type(16))) _Float16 v16h;
typedef __attribute__((ext_vector_type(8)))  _Float16 v8h;
typedef __attribute__((ext_vector_type(8)))  float     v8f;
typedef __attribute__((ext_vector_type(4)))  float     v4f;

#define FF   128
#define AP   40                  // A LDS row stride in f16 (80 B, 16B-aligned sub-chunks)
#define BP   40                  // B LDS row stride in f16
#define XH_ELEMS  (128ull * 34 * 34 * 64)           // padded f16 x: [img][34][34][64]
#define XH_BYTES  (XH_ELEMS * 2)                    // 18,939,904 (256B multiple)

// ---------------------------------------------------------------------------
// Pass 1a: x (8,16,32,32,64) f32 -> zero-padded f16 halo layout [128][34][34][64]
// 295,936 threads, each converts 32 channels of one padded pixel.
// ---------------------------------------------------------------------------
__global__ __launch_bounds__(256) void convert_x_pad(const float* __restrict__ x,
                                                     h16* __restrict__ xh)
{
    const int t    = blockIdx.x * blockDim.x + threadIdx.x;
    const int half = t & 1;
    const int pix  = t >> 1;                  // img*1156 + ph*34 + pw
    const int img  = pix / 1156;
    const int pp   = pix - img * 1156;
    const int ph   = pp / 34;
    const int pw   = pp - ph * 34;

    h16* dst = xh + (size_t)pix * 64 + half * 32;
    const int ih = ph - 1, iw = pw - 1;
    if ((unsigned)ih < 32u && (unsigned)iw < 32u) {
        const float* src = x + (((size_t)img * 32 + ih) * 32 + iw) * 64 + half * 32;
#pragma unroll
        for (int g = 0; g < 4; ++g) {
            v4f f0 = *(const v4f*)(src + g * 8);
            v4f f1 = *(const v4f*)(src + g * 8 + 4);
            v8h h;
#pragma unroll
            for (int j = 0; j < 4; ++j) { h[j] = (h16)f0[j]; h[j + 4] = (h16)f1[j]; }
            *(v8h*)(dst + g * 8) = h;
        }
    } else {
        v8h z = {};
#pragma unroll
        for (int g = 0; g < 4; ++g) *(v8h*)(dst + g * 8) = z;
    }
}

// ---------------------------------------------------------------------------
// Pass 1b: kernel (8,3,3,64,128) f32 -> transposed f16 [b][tap][f][c]
// 9216 threads, one (b,tap,f) row of 64 channels each.
// ---------------------------------------------------------------------------
__global__ __launch_bounds__(256) void convert_w_T(const float* __restrict__ w,
                                                   h16* __restrict__ wh)
{
    const int t  = blockIdx.x * blockDim.x + threadIdx.x;   // (b*9+tap)*128 + f
    const int f  = t & 127;
    const int bt = t >> 7;
    const float* src = w + (size_t)bt * 64 * 128 + f;
    h16* dst = wh + (size_t)t * 64;
#pragma unroll
    for (int g = 0; g < 8; ++g) {
        v8h h;
#pragma unroll
        for (int j = 0; j < 8; ++j)
            h[j] = (h16)src[(size_t)(g * 8 + j) * 128];
        *(v8h*)(dst + g * 8) = h;
    }
}

// ---------------------------------------------------------------------------
// Pass 2: GEMM.  One block = 256 threads = 8 wave32, computes a 128(d) x 128(f)
// tile.  K = 576 in 18 steps of 32.  A/B tiles double-buffered in LDS, filled
// with global_load_async_to_lds_b128 (ASYNCcnt), consumed by
// v_wmma_f32_16x16x32_f16.
// ---------------------------------------------------------------------------
__launch_bounds__(256)
__global__ void conv_wmma_kernel(const h16* __restrict__ xh,
                                 const h16* __restrict__ wh,
                                 float* __restrict__ out)
{
    __shared__ __align__(16) h16 Alds[2][128 * AP];
    __shared__ __align__(16) h16 Blds[2][128 * BP];

    const int tid  = threadIdx.x;
    const int lane = tid & 31;
    const int wave = tid >> 5;
    const int wm   = wave & 3;            // 4 waves along M (32 rows each)
    const int wn   = wave >> 2;           // 2 waves along N (64 cols each)

    const int bid  = blockIdx.x;
    const int dblk = bid & 7;
    const int img  = (bid >> 3) & 15;
    const int bb   = bid >> 7;

    const int    dBase  = dblk * 128;
    const size_t xImgP  = (size_t)(bb * 16 + img) * 34 * 34 * 64;
    const size_t wBaseT = (size_t)bb * 9 * 128 * 64;
    const size_t oBase  = (size_t)(bb * 16 + img) * 1024 * FF;

    const unsigned aB[2] = { (unsigned)(size_t)&Alds[0][0], (unsigned)(size_t)&Alds[1][0] };
    const unsigned bB[2] = { (unsigned)(size_t)&Blds[0][0], (unsigned)(size_t)&Blds[1][0] };

    v8f acc[2][4];
#pragma unroll
    for (int m = 0; m < 2; ++m)
#pragma unroll
        for (int n = 0; n < 4; ++n)
            acc[m][n] = (v8f){0.f, 0.f, 0.f, 0.f, 0.f, 0.f, 0.f, 0.f};

    // Issue the async fills for K-step s into buffer buf.
    // 512 16-byte chunks per tile; thread handles chunks {tid, tid+256} of each.
    auto issue = [&](int s, int buf) {
        const int kh  = s / 6;
        const int kw  = (s >> 1) % 3;
        const int c0  = (s & 1) * 32;
        const size_t whTap = wBaseT + (size_t)((kh * 3 + kw) * 128) * 64;
#pragma unroll
        for (int piece = 0; piece < 2; ++piece) {
            const int ci = tid + piece * 256;
            const int r  = ci >> 2;          // A row (=d-local) / B row (=f)
            const int q  = ci & 3;           // 8-f16 sub-chunk
            // ---- A: xh[img][dblk*4 + r/32 + kh][r%32 + kw][c0 + q*8 ..] ----
            const int ih = dblk * 4 + (r >> 5) + kh;   // padded coords, no bounds test
            const int iw = (r & 31) + kw;
            const unsigned long long ga = (unsigned long long)(xh + xImgP +
                (size_t)(ih * 34 + iw) * 64 + c0 + q * 8);
            const unsigned la = aB[buf] + (unsigned)((r * AP + q * 8) * 2);
            asm volatile("global_load_async_to_lds_b128 %0, %1, off"
                         :: "v"(la), "v"(ga) : "memory");
            // ---- B: wh[b][tap][f=r][c0 + q*8 ..] ----
            const unsigned long long gb = (unsigned long long)(wh + whTap +
                (size_t)r * 64 + c0 + q * 8);
            const unsigned lb = bB[buf] + (unsigned)((r * BP + q * 8) * 2);
            asm volatile("global_load_async_to_lds_b128 %0, %1, off"
                         :: "v"(lb), "v"(gb) : "memory");
        }
    };

    issue(0, 0);                      // prologue fill of buffer 0

    const int hb   = (lane >> 4) * 8;     // A K sub-offset per half-wave
    const int bkof = (lane >> 4) * 8;     // B K sub-offset per half-wave
    const int l15  = lane & 15;

    for (int step = 0; step < 18; ++step) {
        const int cur = step & 1;

        asm volatile("s_wait_asynccnt 0x0" ::: "memory");  // my fills for buf cur done
        __syncthreads();                                   // everyone's fills visible

        if (step < 17)
            issue(step + 1, cur ^ 1);                      // overlap with WMMAs below

        const h16* Ab = &Alds[cur][0];
        const h16* Bb = &Blds[cur][0];
#pragma unroll
        for (int mt = 0; mt < 2; ++mt) {
            const int arow = wm * 32 + mt * 16 + l15;
            const h16* ap = Ab + arow * AP;
            v8h a0 = *(const v8h*)(ap + hb);        // K = hb .. hb+7
            v8h a1 = *(const v8h*)(ap + hb + 16);   // K = hb+16 .. hb+23
            v16h afrag = __builtin_shufflevector(a0, a1,
                0, 1, 2, 3, 4, 5, 6, 7, 8, 9, 10, 11, 12, 13, 14, 15);
#pragma unroll
            for (int nt = 0; nt < 4; ++nt) {
                const int bcol = wn * 64 + nt * 16 + l15;
                const h16* bp = Bb + bcol * BP + bkof;
                v8h b0 = *(const v8h*)(bp);
                v8h b1 = *(const v8h*)(bp + 16);
                v16h bfrag = __builtin_shufflevector(b0, b1,
                    0, 1, 2, 3, 4, 5, 6, 7, 8, 9, 10, 11, 12, 13, 14, 15);
                acc[mt][nt] = __builtin_amdgcn_wmma_f32_16x16x32_f16(
                    false, afrag, false, bfrag,
                    (short)0, acc[mt][nt], false, false);
            }
        }
    }

    // ---- store C (f32, [d][f]) ----
    const int nb = lane & 15;
    const int mb = (lane >> 4) * 8;
#pragma unroll
    for (int mt = 0; mt < 2; ++mt) {
#pragma unroll
        for (int nt = 0; nt < 4; ++nt) {
            const int row0 = dBase + wm * 32 + mt * 16 + mb;
            const int col  = wn * 64 + nt * 16 + nb;
            float* o = out + oBase + (size_t)row0 * FF + col;
#pragma unroll
            for (int v = 0; v < 8; ++v)
                o[(size_t)v * FF] = acc[mt][nt][v];
        }
    }
}

extern "C" void kernel_launch(void* const* d_in, const int* in_sizes, int n_in,
                              void* d_out, int out_size, void* d_ws, size_t ws_size,
                              hipStream_t stream) {
    const float* x   = (const float*)d_in[0];   // (8,16,32,32,64) f32
    const float* w   = (const float*)d_in[1];   // (8,3,3,64,128) f32
    float*       out = (float*)d_out;           // (8,16,32,32,128) f32

    h16* xh = (h16*)d_ws;                                   // 18.94 MB padded f16 x
    h16* wh = (h16*)((char*)d_ws + XH_BYTES);               // 1.18 MB transposed f16 w

    hipLaunchKernelGGL(convert_x_pad,   dim3(1156), dim3(256), 0, stream, x, xh);
    hipLaunchKernelGGL(convert_w_T,     dim3(36),   dim3(256), 0, stream, w, wh);
    hipLaunchKernelGGL(conv_wmma_kernel, dim3(1024), dim3(256), 0, stream, xh, wh, out);
}
